// SimpleAttention_18734647345730
// MI455X (gfx1250) — matrix operand order repaired and checked by reading, not compiled, not run on hardware
//
#include <hip/hip_runtime.h>

typedef __attribute__((ext_vector_type(16))) _Float16 v16h;
typedef __attribute__((ext_vector_type(8)))  _Float16 v8h;
typedef __attribute__((ext_vector_type(8)))  float    v8f;

constexpr int Bn  = 4;
constexpr int Sn  = 2048;
constexpr int Dn  = 1024;   // D_IN == D_OUT
constexpr int BSn = Bn * Sn;
constexpr int QT  = 64;     // query rows per attention block (P strip = 256 KB LDS)

// ---------------------------------------------------------------------------
// WMMA helper: D = A(16x32 f16) * B(32x16 f16) + C(16x16 f32)
// ---------------------------------------------------------------------------
__device__ __forceinline__ v8f wmma16x16x32(v16h a, v16h b, v8f c) {
  return __builtin_amdgcn_wmma_f32_16x16x32_f16(
      /*neg_a=*/false, a, /*neg_b=*/false, b,
      /*c_mod=*/(short)0, c, /*reuse_a=*/false, /*reuse_b=*/false);
}

// A-fragment loader for row-major A with row pointer `rowptr` (already offset
// by this lane's row = lane%16).  Per ISA: lane L holds K = {8*hi..8*hi+7,
// 16+8*hi..16+8*hi+7} where hi = L/16.  Two contiguous 16-byte chunks.
__device__ __forceinline__ v16h load_a_frag(const _Float16* rowptr, int k0, int hi) {
  v8h lo  = *(const v8h*)(rowptr + k0 + hi * 8);
  v8h hi8 = *(const v8h*)(rowptr + k0 + hi * 8 + 16);
  return __builtin_shufflevector(lo, hi8, 0, 1, 2, 3, 4, 5, 6, 7,
                                          8, 9, 10, 11, 12, 13, 14, 15);
}

// ---------------------------------------------------------------------------
// f32 -> f16 conversion (n must be a multiple of 256)
// ---------------------------------------------------------------------------
__global__ void cvt_f32_to_f16(const float* __restrict__ src,
                               _Float16* __restrict__ dst, int n) {
  int i = blockIdx.x * blockDim.x + threadIdx.x;
  if (i < n) dst[i] = (_Float16)src[i];
}

// ---------------------------------------------------------------------------
// QKV projection GEMM.  Block = 8 waves; each wave computes a 16(M) x 128(N)
// tile of one of Q/K/V (selected by blockIdx.z).  Weights are already
// contraction-major [d][e], matching the WMMA B-fragment layout (lane =
// N column, 16 elements over contraction).
// Outputs: q row-major [BS][D], v row-major [BS][D], k TRANSPOSED [b][e][s]
// (so k serves directly as the contraction-major B of the Q*K^T GEMM).
// ---------------------------------------------------------------------------
__global__ __launch_bounds__(256) void qkv_gemm(
    const _Float16* __restrict__ xh,
    const _Float16* __restrict__ wq,
    const _Float16* __restrict__ wk,
    const _Float16* __restrict__ wv,
    const float* __restrict__ bq,
    const float* __restrict__ bk,
    const float* __restrict__ bv,
    _Float16* __restrict__ qh,
    _Float16* __restrict__ kTh,
    _Float16* __restrict__ vh) {
  const int lane = threadIdx.x & 31;
  const int wave = threadIdx.x >> 5;
  const int row  = lane & 15;
  const int hi   = lane >> 4;

  const int sel = blockIdx.z;                 // 0=Q, 1=K, 2=V
  const int n0  = blockIdx.x * 128;
  const int m0  = blockIdx.y * 128 + wave * 16;

  const _Float16* W    = (sel == 0) ? wq : (sel == 1) ? wk : wv;
  const float*    bias = (sel == 0) ? bq : (sel == 1) ? bk : bv;

  const _Float16* arow = xh + (size_t)(m0 + row) * Dn;

  v8f acc[8] = {};
  for (int k0 = 0; k0 < Dn; k0 += 32) {
    v16h a = load_a_frag(arow, k0, hi);
    const _Float16* brow = W + (size_t)k0 * Dn + n0 + row;  // lane%16 = N column, elements over K
#pragma unroll
    for (int j = 0; j < 8; ++j) {
      v16h bfrag; for (int s = 0; s < 16; ++s) bfrag[s] = brow[(size_t)((s & 7) + 8 * hi + 16 * (s >> 3)) * Dn + 16 * j];
      acc[j] = wmma16x16x32(a, bfrag, acc[j]);
    }
  }

  // C/D layout: element (M = t + 8*hi, N = n0 + 16j + row) lives in acc[j][t].
#pragma unroll
  for (int j = 0; j < 8; ++j) {
    const int   n  = n0 + 16 * j + row;
    const float bb = bias[n];
#pragma unroll
    for (int t = 0; t < 8; ++t) {
      const int      m   = m0 + t + 8 * hi;
      const _Float16 val = (_Float16)(acc[j][t] + bb);
      if (sel == 1) {                       // K: store transposed [b][e][s]
        const int bidx = m >> 11;           // m / Sn
        const int s    = m & (Sn - 1);      // m % Sn
        kTh[(size_t)bidx * Dn * Sn + (size_t)n * Sn + s] = val;
      } else {
        _Float16* dst = (sel == 0) ? qh : vh;
        dst[(size_t)m * Dn + n] = val;
      }
    }
  }
}

// ---------------------------------------------------------------------------
// Attention: one block (16 waves, 512 threads) per 64-row query tile.
// Phase 1: scores strip (64 x S) = Q_tile * K^T (scaled) -> LDS f16 (256 KB).
//          Wave w handles row-tile (w&3), kseq slice (w>>2).
// Phase 2: row softmax, 8 threads/row, shfl_xor reductions in 8-lane groups,
//          probabilities normalized in place.
// Phase 3: wave w computes row-tile (w&3) x 256 output columns (w>>2) of
//          O = P * V (16 accumulators), stores f32.
// ---------------------------------------------------------------------------
__global__ __launch_bounds__(512) void attn_kernel(
    const _Float16* __restrict__ qh,
    const _Float16* __restrict__ kTh,
    const _Float16* __restrict__ vh,
    float* __restrict__ out) {
  __shared__ _Float16 sP[QT][Sn];           // 256 KB

  const int tid  = threadIdx.x;
  const int lane = tid & 31;
  const int wave = tid >> 5;                // 0..15
  const int row  = lane & 15;
  const int hi   = lane >> 4;

  const int qt = blockIdx.x;                // 0 .. B*S/QT - 1
  const int b  = qt / (Sn / QT);
  const int s0 = (qt % (Sn / QT)) * QT;

  const _Float16* kTb = kTh + (size_t)b * Dn * Sn;
  const _Float16* vb  = vh + (size_t)b * Sn * Dn;

  constexpr float scale = 0.03125f;         // 1/sqrt(1024)

  // ---- Phase 1: scores -> LDS -------------------------------------------
  {
    const int rw     = wave & 3;            // row tile
    const int kslice = wave >> 2;           // kseq slice
    const _Float16* arow = qh + ((size_t)b * Sn + s0 + rw * 16 + row) * Dn;
    for (int t = kslice * 32; t < kslice * 32 + 32; ++t) {
      const int ks0 = t * 16;
      v8f c = {};
      for (int e0 = 0; e0 < Dn; e0 += 32) {
        v16h a = load_a_frag(arow, e0, hi);
        v16h bfrag; for (int s = 0; s < 16; ++s) bfrag[s] = kTb[(size_t)(e0 + (s & 7) + 8 * hi + 16 * (s >> 3)) * Sn + ks0 + row];
        c = wmma16x16x32(a, bfrag, c);
      }
#pragma unroll
      for (int tt = 0; tt < 8; ++tt)
        sP[rw * 16 + tt + 8 * hi][ks0 + row] = (_Float16)(c[tt] * scale);
    }
  }
  __syncthreads();

  // ---- Phase 2: softmax, normalize P in place ---------------------------
  {
    const int r   = tid >> 3;               // 0..63, row of the strip
    const int sub = tid & 7;                // 8 threads per row (aligned group)
    const int c0  = sub * (Sn / 8);         // 256-column chunk
    float mx = -3.0e38f;
    for (int cix = c0; cix < c0 + Sn / 8; ++cix)
      mx = fmaxf(mx, (float)sP[r][cix]);
#pragma unroll
    for (int off = 1; off < 8; off <<= 1)   // masks <8 stay in 8-lane group
      mx = fmaxf(mx, __shfl_xor(mx, off, 32));
    float sm = 0.f;
    for (int cix = c0; cix < c0 + Sn / 8; ++cix) {
      const float p = __expf((float)sP[r][cix] - mx);
      sP[r][cix]    = (_Float16)p;
      sm += p;
    }
#pragma unroll
    for (int off = 1; off < 8; off <<= 1)
      sm += __shfl_xor(sm, off, 32);
    const float inv = 1.0f / sm;
    for (int cix = c0; cix < c0 + Sn / 8; ++cix)
      sP[r][cix] = (_Float16)((float)sP[r][cix] * inv);
  }
  __syncthreads();

  // ---- Phase 3: O = P * V -----------------------------------------------
  {
    const int rw = wave & 3;                // row tile
    const int e0 = (wave >> 2) * 256;       // column group
    const _Float16* prow = &sP[rw * 16 + row][0];
    v8f acc[16] = {};
    for (int ks = 0; ks < Sn; ks += 32) {
      v16h a = load_a_frag(prow, ks, hi);
      const _Float16* brow = vb + (size_t)ks * Dn + e0 + row;  // lane%16 = N column, elements over K
#pragma unroll
      for (int j = 0; j < 16; ++j) {
        v16h bfrag; for (int s = 0; s < 16; ++s) bfrag[s] = brow[(size_t)((s & 7) + 8 * hi + 16 * (s >> 3)) * Dn + 16 * j];
        acc[j] = wmma16x16x32(a, bfrag, acc[j]);
      }
    }
    const size_t mbase = (size_t)b * Sn + s0 + rw * 16;
#pragma unroll
    for (int j = 0; j < 16; ++j) {
      const int n = e0 + 16 * j + row;
#pragma unroll
      for (int t = 0; t < 8; ++t)
        out[(mbase + t + 8 * hi) * Dn + n] = (float)acc[j][t];
    }
  }
}

// ---------------------------------------------------------------------------
// Launch
// ---------------------------------------------------------------------------
extern "C" void kernel_launch(void* const* d_in, const int* in_sizes, int n_in,
                              void* d_out, int out_size, void* d_ws, size_t ws_size,
                              hipStream_t stream) {
  const float* x  = (const float*)d_in[0];
  const float* Wq = (const float*)d_in[1];
  const float* bq = (const float*)d_in[2];
  const float* Wk = (const float*)d_in[3];
  const float* bk = (const float*)d_in[4];
  const float* Wv = (const float*)d_in[5];
  const float* bv = (const float*)d_in[6];
  float* out = (float*)d_out;

  // Workspace layout (f16 elements).  Total = 4*BS*D + 3*D*D halves ~ 73 MB.
  const int NX = BSn * Dn;   // 8,388,608
  const int NW = Dn * Dn;    // 1,048,576
  _Float16* ws  = (_Float16*)d_ws;
  _Float16* xh  = ws;
  _Float16* wqh = xh + (size_t)NX;
  _Float16* wkh = wqh + (size_t)NW;
  _Float16* wvh = wkh + (size_t)NW;
  _Float16* qh  = wvh + (size_t)NW;
  _Float16* kTh = qh + (size_t)NX;
  _Float16* vh  = kTh + (size_t)NX;

  cvt_f32_to_f16<<<NX / 256, 256, 0, stream>>>(x, xh, NX);
  cvt_f32_to_f16<<<NW / 256, 256, 0, stream>>>(Wq, wqh, NW);
  cvt_f32_to_f16<<<NW / 256, 256, 0, stream>>>(Wk, wkh, NW);
  cvt_f32_to_f16<<<NW / 256, 256, 0, stream>>>(Wv, wvh, NW);

  qkv_gemm<<<dim3(Dn / 128, BSn / 128, 3), 256, 0, stream>>>(
      xh, wqh, wkh, wvh, bq, bk, bv, qh, kTh, vh);

  attn_kernel<<<dim3(BSn / QT), 512, 0, stream>>>(qh, kTh, vh, out);
}
